// VectorQuantizerEMA_49435073577317
// MI455X (gfx1250) — matrix operand order repaired and checked by reading, not compiled, not run on hardware
//
#include <hip/hip_runtime.h>

// ---------------------------------------------------------------------------
// VQ-VAE EMA vector quantizer for MI455X (gfx1250, wave32, WMMA).
//   N = B*T = 16384 rows, D = 256, K = 8192 codes.
// Distance search = f16 WMMA GEMM (f32 accum), B staged via async-copy to
// LDS (double buffered), fused row-argmin, K split 2-way for occupancy.
// ---------------------------------------------------------------------------

#define NROWS   16384          // B*T
#define DIM     256
#define KCODES  8192
#define KHALF   (KCODES / 2)
#define GAMMA_F 0.99f
#define OMG_F   0.01f          // 1 - gamma
#define COMMIT  0.25f

typedef __attribute__((ext_vector_type(16))) _Float16 v16h;
typedef __attribute__((ext_vector_type(8)))  _Float16 v8h;
typedef __attribute__((ext_vector_type(8)))  float    v8f;
typedef __attribute__((ext_vector_type(4)))  int      v4i;

// 16-byte global -> LDS copy. gfx1250 async path (ASYNCcnt, overlaps with
// WMMA). Builtin signature (from toolchain): (v4i AS1*, v4i AS3*, Ii, Ii).
__device__ __forceinline__ void async_cp16(const void* g, void* l) {
#if __has_builtin(__builtin_amdgcn_global_load_async_to_lds_b128)
    __builtin_amdgcn_global_load_async_to_lds_b128(
        (__attribute__((address_space(1))) v4i*)(v4i*)g,
        (__attribute__((address_space(3))) v4i*)(v4i*)l,
        0, 0);
#else
    *(uint4*)l = *(const uint4*)g;
#endif
}

__device__ __forceinline__ void wait_async0() {
#if __has_builtin(__builtin_amdgcn_s_wait_asynccnt)
    __builtin_amdgcn_s_wait_asynccnt(0);
#else
    asm volatile("s_wait_asynccnt 0x0" ::: "memory");
#endif
}

// --------------------------- fp32 -> f16 copy of x -------------------------
__global__ void __launch_bounds__(256)
vq_convert_x(const float* __restrict__ x, _Float16* __restrict__ xh) {
    int i = blockIdx.x * 256 + threadIdx.x;          // exactly N*D threads
    xh[i] = (_Float16)x[i];
}

// -------- fp32 -> f16 codebook scaled by -2, plus ||e||^2 per code ---------
__global__ void __launch_bounds__(256)
vq_convert_e(const float* __restrict__ e, _Float16* __restrict__ eh,
             float* __restrict__ e_sq) {
    __shared__ float red[8];
    int k = blockIdx.x;                               // one block per code row
    int t = threadIdx.x;                              // t in [0,256) == d
    float v = e[k * DIM + t];
    eh[k * DIM + t] = (_Float16)(-2.0f * v);          // fold the -2 of -2*x.e
    float s = v * v;
    #pragma unroll
    for (int m = 16; m >= 1; m >>= 1) s += __shfl_xor(s, m, 32);
    if ((t & 31) == 0) red[t >> 5] = s;
    __syncthreads();
    if (t == 0) {
        float tot = 0.f;
        #pragma unroll
        for (int w = 0; w < 8; ++w) tot += red[w];
        e_sq[k] = tot;
    }
}

// ---------------- fused distance-GEMM + argmin (WMMA f16/f32) --------------
// Grid: 256 blocks x 256 threads (8 waves). Block (mgroup, kq) covers 8
// M-tiles x one K half. Each wave owns one 16-row M tile with the full
// 16x256 A panel in registers. B tiles (32 codes/iter) are staged into LDS
// with async copies, double buffered; 4 independent WMMA accumulator chains
// keep the XDL pipe busy. ||x||^2 omitted (constant per row under argmin).
__global__ void __launch_bounds__(256)
vq_argmin(const _Float16* __restrict__ xh, const _Float16* __restrict__ eh,
          const float* __restrict__ e_sq,
          float* __restrict__ pminv, int* __restrict__ pmini) {
    __shared__ alignas(32) _Float16 Bt[2][32 * DIM];  // 2 x 16 KB
    __shared__ float esl[KHALF];                      // 16 KB

    const int lane   = threadIdx.x & 31;
    const int wave   = threadIdx.x >> 5;
    const int kq     = blockIdx.x & 1;
    const int mgroup = blockIdx.x >> 1;
    const int mtile  = mgroup * 8 + wave;             // 0..1023
    const int half   = lane >> 4;                     // 0: lanes 0-15, 1: 16-31
    const int lrow   = lane & 15;
    const int cbase  = kq * KHALF;

    // stage ||e||^2 for our K half into LDS (one-time)
    for (int i = threadIdx.x; i < KHALF; i += 256)
        esl[i] = e_sq[cbase + i];

    // ---- A panel: 8 fragments of 16x32 f16 covering D = 256 ----
    // Layout: lane holds row M=lane&15; v16h = [k: h*16..h*16+15]
    //         (h = lane>=16), per 32-wide d chunk (matches B fragments).
    const _Float16* xrow = xh + (size_t)(mtile * 16 + lrow) * DIM;
    v16h A[8];
    #pragma unroll
    for (int c = 0; c < 8; ++c) {
        const int d0 = c * 32;
        v8h lo = *(const v8h*)(xrow + d0 + half * 16);
        v8h hi = *(const v8h*)(xrow + d0 + half * 16 + 8);
        A[c] = __builtin_shufflevector(lo, hi, 0, 1, 2, 3, 4, 5, 6, 7,
                                               8, 9, 10, 11, 12, 13, 14, 15);
    }

    // B staging: iteration = 32 code rows = 16384 contiguous bytes of eh.
    // 8 waves x 32 lanes x 4 chunks x 16 B = 16384 B.
    const char* gbase = (const char*)eh + (size_t)cbase * (DIM * 2);
    auto stage = [&](int it, int buf) {
        const char* g = gbase + (size_t)it * 16384 + wave * 2048 + lane * 16;
        char*       l = (char*)&Bt[buf][0] + wave * 2048 + lane * 16;
        #pragma unroll
        for (int j = 0; j < 4; ++j)
            async_cp16(g + j * 512, l + j * 512);
    };

    stage(0, 0);
    wait_async0();
    __syncthreads();

    float minval[8];
    int   minidx[8];
    #pragma unroll
    for (int r = 0; r < 8; ++r) { minval[r] = __builtin_inff(); minidx[r] = 0; }

    const int NIT = KHALF / 32;                       // 128 iterations
    for (int it = 0; it < NIT; ++it) {
        const int buf = it & 1;
        if (it + 1 < NIT) stage(it + 1, buf ^ 1);     // prefetch overlaps WMMA

        // B fragments from LDS: lane holds code row (tile + lane&15),
        // contiguous 16 halves at d0 + 16*half.
        const _Float16* b0 = &Bt[buf][(lrow)      * DIM + half * 16];
        const _Float16* b1 = &Bt[buf][(16 + lrow) * DIM + half * 16];

        v8f a00 = {}, a01 = {}, a10 = {}, a11 = {};   // 4 independent chains
        #pragma unroll
        for (int c = 0; c < 4; ++c) {
            v16h B00 = *(const v16h*)(b0 + c * 32);
            v16h B10 = *(const v16h*)(b1 + c * 32);
            v16h B01 = *(const v16h*)(b0 + (c + 4) * 32);
            v16h B11 = *(const v16h*)(b1 + (c + 4) * 32);
            a00 = __builtin_amdgcn_wmma_f32_16x16x32_f16(
                false, A[c],     false, B00, (short)0, a00, false, false);
            a10 = __builtin_amdgcn_wmma_f32_16x16x32_f16(
                false, A[c],     false, B10, (short)0, a10, false, false);
            a01 = __builtin_amdgcn_wmma_f32_16x16x32_f16(
                false, A[c + 4], false, B01, (short)0, a01, false, false);
            a11 = __builtin_amdgcn_wmma_f32_16x16x32_f16(
                false, A[c + 4], false, B11, (short)0, a11, false, false);
        }

        // C layout: VGPR r on lane n = (M = 8*half + r, N = n&15).
        // eh was pre-scaled by -2, so dist = (-2 x.e) + ||e||^2.
        const int   code0 = it * 32 + lrow;
        const int   code1 = code0 + 16;
        const float e0 = esl[code0];
        const float e1 = esl[code1];
        #pragma unroll
        for (int r = 0; r < 8; ++r) {
            float d0 = a00[r] + a01[r] + e0;
            float d1 = a10[r] + a11[r] + e1;
            if (d0 < minval[r]) { minval[r] = d0; minidx[r] = code0; }
            if (d1 < minval[r]) { minval[r] = d1; minidx[r] = code1; }
        }

        wait_async0();                                // our prefetch landed
        __syncthreads();                              // whole block in step
    }

    // Cross-lane argmin over the 16 column slots (xor masks stay inside each
    // 16-lane half). Tie-break toward the smaller code index.
    #pragma unroll
    for (int r = 0; r < 8; ++r) {
        float v = minval[r];
        int   i = minidx[r];
        #pragma unroll
        for (int m = 8; m >= 1; m >>= 1) {
            float ov = __shfl_xor(v, m, 32);
            int   oi = __shfl_xor(i, m, 32);
            if (ov < v || (ov == v && oi < i)) { v = ov; i = oi; }
        }
        if (lrow == 0) {
            int row = mtile * 16 + half * 8 + r;
            pminv[kq * NROWS + row] = v;
            pmini[kq * NROWS + row] = cbase + i;
        }
    }
}

// -------------------- merge the two K-half partial argmins -----------------
__global__ void __launch_bounds__(256)
vq_combine(const float* __restrict__ pminv, const int* __restrict__ pmini,
           int* __restrict__ idx_out) {
    int r = blockIdx.x * 256 + threadIdx.x;
    float v0 = pminv[r];         int i0 = pmini[r];
    float v1 = pminv[NROWS + r]; int i1 = pmini[NROWS + r];
    idx_out[r] = (v1 < v0 || (v1 == v0 && i1 < i0)) ? i1 : i0;
}

// ------ gather quantized, scatter dw/counts, commitment-loss partials ------
__global__ void __launch_bounds__(256)
vq_scatter(const float* __restrict__ x, const float* __restrict__ emb,
           const int* __restrict__ idx, float* __restrict__ out_q,
           float* __restrict__ dw, float* __restrict__ counts,
           float* __restrict__ loss_acc) {
    __shared__ float red[8];
    int n = blockIdx.x;                               // one block per row
    int t = threadIdx.x;                              // d index
    int code = idx[n];
    float q  = emb[code * DIM + t];
    float xv = x[(size_t)n * DIM + t];
    out_q[(size_t)n * DIM + t] = q;                   // quantized_st == quantized
    atomicAdd(&dw[code * DIM + t], xv);               // segment_sum(flat_x, idx)
    float d = xv - q;
    float s = d * d;
    #pragma unroll
    for (int m = 16; m >= 1; m >>= 1) s += __shfl_xor(s, m, 32);
    if ((t & 31) == 0) red[t >> 5] = s;
    __syncthreads();
    if (t == 0) {
        float tot = 0.f;
        #pragma unroll
        for (int w = 0; w < 8; ++w) tot += red[w];
        atomicAdd(loss_acc, tot);
        atomicAdd(&counts[code], 1.0f);               // segment_sum(ones, idx)
    }
}

// ---------------- cluster-size EMA + global sum n = sum(ncs) ---------------
__global__ void __launch_bounds__(256)
vq_cluster(const float* __restrict__ ema_cs, const float* __restrict__ counts,
           float* __restrict__ out_cs, float* __restrict__ n_acc) {
    __shared__ float red[8];
    int k = blockIdx.x * 256 + threadIdx.x;
    float c   = counts[k];
    float enc = (c > 0.f) ? c : 1.f;
    float ncs = GAMMA_F * ema_cs[k] + OMG_F * enc;
    out_cs[k] = ncs;
    float s = ncs;
    #pragma unroll
    for (int m = 16; m >= 1; m >>= 1) s += __shfl_xor(s, m, 32);
    if ((threadIdx.x & 31) == 0) red[threadIdx.x >> 5] = s;
    __syncthreads();
    if (threadIdx.x == 0) {
        float tot = 0.f;
        #pragma unroll
        for (int w = 0; w < 8; ++w) tot += red[w];
        atomicAdd(n_acc, tot);
    }
}

// ------------- embedding EMA, normalized weights, loss scalar --------------
__global__ void __launch_bounds__(256)
vq_finalize(const float* __restrict__ ema_emb, const float* __restrict__ dw,
            const float* __restrict__ out_cs, const float* __restrict__ n_acc,
            const float* __restrict__ loss_acc,
            float* __restrict__ out_w, float* __restrict__ out_ema,
            float* __restrict__ out_loss) {
    int i = blockIdx.x * 256 + threadIdx.x;           // K*D threads
    int k = i >> 8;                                   // i / DIM
    float nee = GAMMA_F * ema_emb[i] + OMG_F * dw[i];
    out_ema[i] = nee;
    out_w[i] = nee * n_acc[0] / out_cs[k];            // nee / (ncs / n)
    if (i == 0)
        out_loss[0] = COMMIT * loss_acc[0] / (float)((size_t)NROWS * DIM);
}

// ---------------------------------------------------------------------------
extern "C" void kernel_launch(void* const* d_in, const int* in_sizes, int n_in,
                              void* d_out, int out_size, void* d_ws, size_t ws_size,
                              hipStream_t stream) {
    const float* x       = (const float*)d_in[0];   // [16,1024,256]
    const float* emb     = (const float*)d_in[1];   // [8192,256]
    const float* ema_cs  = (const float*)d_in[2];   // [8192]
    const float* ema_emb = (const float*)d_in[3];   // [8192,256]

    float* out      = (float*)d_out;
    float* out_q    = out;                          // 4,194,304
    float* out_loss = out + 4194304;                // 1
    float* out_w    = out + 4194305;                // 2,097,152
    float* out_cs   = out + 6291457;                // 8,192
    float* out_ema  = out + 6299649;                // 2,097,152

    // workspace layout (~21.3 MB)
    char* ws = (char*)d_ws;
    _Float16* xh     = (_Float16*)(ws);                                // 8 MB
    _Float16* eh     = (_Float16*)(ws + (8u  << 20));                  // 4 MB
    float*    e_sq   = (float*)   (ws + (12u << 20));                  // 32 KB
    int*      idx    = (int*)     (ws + (12u << 20) + (64u << 10));    // 64 KB
    float*    counts = (float*)   (ws + (12u << 20) + (128u << 10));   // 32 KB
    float*    dw     = (float*)   (ws + (13u << 20));                  // 8 MB
    float*    scal   = (float*)   (ws + (21u << 20));                  // [loss, n]
    float*    pminv  = (float*)   (ws + (21u << 20) + (4u  << 10));    // 128 KB
    int*      pmini  = (int*)     (ws + (21u << 20) + (164u << 10));   // 128 KB

    (void)hipMemsetAsync(counts, 0, KCODES * sizeof(float), stream);
    (void)hipMemsetAsync(dw,     0, (size_t)KCODES * DIM * sizeof(float), stream);
    (void)hipMemsetAsync(scal,   0, 2 * sizeof(float), stream);

    vq_convert_x<<<(NROWS * DIM) / 256, 256, 0, stream>>>(x, xh);
    vq_convert_e<<<KCODES, 256, 0, stream>>>(emb, eh, e_sq);
    vq_argmin  <<<256, 256, 0, stream>>>(xh, eh, e_sq, pminv, pmini);
    vq_combine <<<NROWS / 256, 256, 0, stream>>>(pminv, pmini, idx);
    vq_scatter <<<NROWS, 256, 0, stream>>>(x, emb, idx, out_q, dw, counts,
                                           scal + 0);
    vq_cluster <<<KCODES / 256, 256, 0, stream>>>(ema_cs, counts, out_cs,
                                                  scal + 1);
    vq_finalize<<<(KCODES * DIM) / 256, 256, 0, stream>>>(
        ema_emb, dw, out_cs, scal + 1, scal + 0, out_w, out_ema, out_loss);
}
